// HierarchicalTextClassifyCnnNet_85615878078958
// MI455X (gfx1250) — matrix operand order repaired and checked by reading, not compiled, not run on hardware
//
#include <hip/hip_runtime.h>
#include <hip/hip_bf16.h>
#include <cstdint>

// ---------------- problem constants ----------------
#define BATCH 512
#define SEQ   128
#define EMB   300
#define EP    320            // padded embedding (multiple of 32 -> aligned WMMA K tiles)
#define OC    128
#define K1C   16
#define K2C   64
#define DFEAT 384            // OC * 3
#define XPAD  2048           // guard elements past xh for window over-read on padded t rows

typedef _Float16 half8  __attribute__((ext_vector_type(8)));
typedef _Float16 half16 __attribute__((ext_vector_type(16)));
typedef float    float8 __attribute__((ext_vector_type(8)));

// Assemble a v16h WMMA fragment from two contiguous 8-halfword (16B) runs.
__device__ __forceinline__ half16 make_frag(const _Float16* p_lo, const _Float16* p_hi) {
    half8 lo = *(const half8*)p_lo;
    half8 hi = *(const half8*)p_hi;
    half16 r;
#pragma unroll
    for (int i = 0; i < 8; ++i) { r[i] = lo[i]; r[i + 8] = hi[i]; }
    return r;
}

// ---------------- conversion kernels ----------------
__global__ __launch_bounds__(256) void convert_x(const float* __restrict__ x,
                                                 _Float16* __restrict__ xh) {
    int idx = blockIdx.x * blockDim.x + threadIdx.x;
    const int n = BATCH * SEQ * EP;
    if (idx >= n + XPAD) return;
    if (idx >= n) { xh[idx] = (_Float16)0.f; return; }     // zero the guard tail
    int col = idx % EP;
    int row = idx / EP;                                    // b*SEQ + s
    xh[idx] = (col < EMB) ? (_Float16)x[(size_t)row * EMB + col] : (_Float16)0.f;
}

__global__ __launch_bounds__(256) void convert_w(const float* __restrict__ w,
                                                 _Float16* __restrict__ wh, int fs) {
    int idx = blockIdx.x * blockDim.x + threadIdx.x;
    const int K = fs * EP;
    if (idx >= OC * K) return;
    int k  = idx % K;
    int oc = idx / K;
    int e  = k % EP;
    int r  = k / EP;
    wh[idx] = (e < EMB) ? (_Float16)w[((size_t)oc * fs + r) * EMB + e] : (_Float16)0.f;
}

// ---------------- fused conv-as-GEMM + bias + ReLU + max-pool ----------------
// grid: (4 oc-tiles, BATCH, 3 filter sizes), block: 128 (4 waves)
// Each wave computes a 32(M=t) x 32(N=oc) tile: 2 A-fragments x 2 B-fragments ->
// 4 WMMAs from 8 b128 loads per k-step (2 VMEM per WMMA).
// Wave w owns t-tiles {w, w+4}; tile w (t<=63) is always in-range, only tile w+4
// needs max-pool masking.
__global__ __launch_bounds__(128) void conv_gemm_max(
    const _Float16* __restrict__ xh,
    const _Float16* __restrict__ wh3, const _Float16* __restrict__ wh4,
    const _Float16* __restrict__ wh5,
    const float* __restrict__ b3, const float* __restrict__ b4,
    const float* __restrict__ b5,
    float* __restrict__ feat) {

    const int ocbase = blockIdx.x * 32;     // 0,32,64,96
    const int b      = blockIdx.y;          // 0..BATCH-1
    const int fsi    = blockIdx.z;          // 0..2
    const int fs     = 3 + fsi;
    const int K      = fs * EP;             // 960 / 1280 / 1600
    const int T      = SEQ - fs + 1;        // 126 / 125 / 124

    const _Float16* wh = (fsi == 0) ? wh3 : (fsi == 1) ? wh4 : wh5;
    const float*    bb = (fsi == 0) ? b3  : (fsi == 1) ? b4  : b5;

    const int wave  = threadIdx.x >> 5;     // 0..3
    const int lane  = threadIdx.x & 31;
    const int lmod  = lane & 15;
    const int lhi   = lane >> 4;            // 0 | 1

    const int tbase0 = wave * 16;           // 0..48   (always < T)
    const int tbase1 = (wave + 4) * 16;     // 64..112 (tail rows need masking)

    // A lane layout (16-bit 16x32): lanes 0-15 hold K {0..7,16..23}, lanes 16-31 {8..15,24..31}
    const _Float16* aptr0 = xh + ((size_t)(b * SEQ + tbase0 + lmod)) * EP + lhi * 8;
    const _Float16* aptr1 = xh + ((size_t)(b * SEQ + tbase1 + lmod)) * EP + lhi * 8;
    // B lane layout (16-bit 32x16): col = lane%16; lanes 0-15 K {0..7,16..23}, lanes 16-31 K {8..15,24..31}
    const _Float16* bptr0 = wh + (size_t)(ocbase + lmod) * K + lhi * 8;
    const _Float16* bptr1 = bptr0 + (size_t)16 * K;

    float8 acc00 = {0.f, 0.f, 0.f, 0.f, 0.f, 0.f, 0.f, 0.f};
    float8 acc01 = acc00;
    float8 acc10 = acc00;
    float8 acc11 = acc00;

#pragma unroll 2
    for (int k0 = 0; k0 < K; k0 += 32) {
        half16 a0 = make_frag(aptr0, aptr0 + 16);       // two 16B runs, 16B gap
        half16 a1 = make_frag(aptr1, aptr1 + 16);
        half16 w0 = make_frag(bptr0, bptr0 + 16);       // two 16B runs, 16B gap
        half16 w1 = make_frag(bptr1, bptr1 + 16);
        acc00 = __builtin_amdgcn_wmma_f32_16x16x32_f16(
            false, a0, false, w0, (short)0, acc00, false, false);
        acc01 = __builtin_amdgcn_wmma_f32_16x16x32_f16(
            false, a0, false, w1, (short)0, acc01, false, false);
        acc10 = __builtin_amdgcn_wmma_f32_16x16x32_f16(
            false, a1, false, w0, (short)0, acc10, false, false);
        acc11 = __builtin_amdgcn_wmma_f32_16x16x32_f16(
            false, a1, false, w1, (short)0, acc11, false, false);
        aptr0 += 32;
        aptr1 += 32;
        bptr0 += 32;
        bptr1 += 32;
    }

    // C layout: VGPR i, lanes 0-15 -> (M=i, N=lane); lanes 16-31 -> (M=8+i, N=lane-16)
    const float bias0 = bb[ocbase + lmod];
    const float bias1 = bb[ocbase + 16 + lmod];
    float m0 = 0.0f, m1 = 0.0f;                          // ReLU => 0 is a safe max identity
#pragma unroll
    for (int i = 0; i < 8; ++i) {
        // M-tile 0: always valid (t <= 63 < T)
        float v00 = acc00[i] + bias0;
        float v01 = acc01[i] + bias1;
        m0 = fmaxf(m0, v00 > 0.f ? v00 : 0.f);
        m1 = fmaxf(m1, v01 > 0.f ? v01 : 0.f);
        // M-tile 1: mask padded rows
        int t1 = tbase1 + i + lhi * 8;
        float v10 = acc10[i] + bias0;
        float v11 = acc11[i] + bias1;
        v10 = v10 > 0.f ? v10 : 0.f;
        v11 = v11 > 0.f ? v11 : 0.f;
        if (t1 < T) { m0 = fmaxf(m0, v10); m1 = fmaxf(m1, v11); }
    }
    m0 = fmaxf(m0, __shfl_xor(m0, 16, 32));              // fuse the two M half-rows per column
    m1 = fmaxf(m1, __shfl_xor(m1, 16, 32));

    __shared__ float s_red[4][32];
    if (lane < 16) {
        s_red[wave][lane]      = m0;
        s_red[wave][lane + 16] = m1;
    }
    __syncthreads();
    if (threadIdx.x < 32) {
        float mm = s_red[0][threadIdx.x];
#pragma unroll
        for (int w = 1; w < 4; ++w) mm = fmaxf(mm, s_red[w][threadIdx.x]);
        feat[(size_t)b * DFEAT + fsi * OC + ocbase + threadIdx.x] = mm;
    }
}

// ---------------- hierarchical softmax head ----------------
__global__ __launch_bounds__(128) void head_kernel(
    const float* __restrict__ feat, const int* __restrict__ targets,
    const float* __restrict__ W_root, const float* __restrict__ b_root,
    const float* __restrict__ W_child, const float* __restrict__ b_child,
    float* __restrict__ out_loss, float* __restrict__ out_root,
    float* __restrict__ out_child) {

    const int b   = blockIdx.x;
    const int tid = threadIdx.x;
    __shared__ float s_feat[DFEAT];
    __shared__ float s_root[K1C];
    __shared__ float s_child[K2C];

    for (int d = tid; d < DFEAT; d += 128) s_feat[d] = feat[(size_t)b * DFEAT + d];
    __syncthreads();

    const int tgt = targets[b];
    const int p0  = tgt / K2C;
    const int p1  = tgt % K2C;

    if (tid < K1C) {
        const float* wr = W_root + (size_t)tid * DFEAT;
        float a = b_root[tid];
        for (int d = 0; d < DFEAT; ++d) a = fmaf(s_feat[d], wr[d], a);
        s_root[tid] = a;
        out_root[b * K1C + tid] = a;
    }
    if (tid < K2C) {
        const float* wc = W_child + ((size_t)(p0 * K2C + tid)) * DFEAT;
        float a = b_child[p0 * K2C + tid];
        for (int d = 0; d < DFEAT; ++d) a = fmaf(s_feat[d], wc[d], a);
        s_child[tid] = a;
        out_child[b * K2C + tid] = a;
    }
    __syncthreads();

    if (tid == 0) {
        float m0 = s_root[0];
        for (int i = 1; i < K1C; ++i) m0 = fmaxf(m0, s_root[i]);
        float z0 = 0.f;
        for (int i = 0; i < K1C; ++i) z0 += __expf(s_root[i] - m0);
        float lse0 = m0 + __logf(z0);

        float m1 = s_child[0];
        for (int i = 1; i < K2C; ++i) m1 = fmaxf(m1, s_child[i]);
        float z1 = 0.f;
        for (int i = 0; i < K2C; ++i) z1 += __expf(s_child[i] - m1);
        float lse1 = m1 + __logf(z1);

        out_loss[b] = (lse0 - s_root[p0]) + (lse1 - s_child[p1]);
    }
}

// ---------------- launcher ----------------
static inline size_t align256(size_t x) { return (x + 255) & ~(size_t)255; }

extern "C" void kernel_launch(void* const* d_in, const int* in_sizes, int n_in,
                              void* d_out, int out_size, void* d_ws, size_t ws_size,
                              hipStream_t stream) {
    const float* x       = (const float*)d_in[0];
    const int*   targets = (const int*)d_in[1];
    const float* W3      = (const float*)d_in[2];
    const float* b3      = (const float*)d_in[3];
    const float* W4      = (const float*)d_in[4];
    const float* b4      = (const float*)d_in[5];
    const float* W5      = (const float*)d_in[6];
    const float* b5      = (const float*)d_in[7];
    const float* W_root  = (const float*)d_in[8];
    const float* b_root  = (const float*)d_in[9];
    const float* W_child = (const float*)d_in[10];
    const float* b_child = (const float*)d_in[11];

    float* out_loss  = (float*)d_out;
    float* out_root  = out_loss + BATCH;
    float* out_child = out_root + (size_t)BATCH * K1C;

    // workspace carve-up
    char* ws = (char*)d_ws;
    _Float16* xh = (_Float16*)ws;
    ws += align256(((size_t)BATCH * SEQ * EP + XPAD) * sizeof(_Float16));
    _Float16* wh3 = (_Float16*)ws; ws += align256((size_t)OC * 3 * EP * sizeof(_Float16));
    _Float16* wh4 = (_Float16*)ws; ws += align256((size_t)OC * 4 * EP * sizeof(_Float16));
    _Float16* wh5 = (_Float16*)ws; ws += align256((size_t)OC * 5 * EP * sizeof(_Float16));
    float*    feat = (float*)ws;   ws += align256((size_t)BATCH * DFEAT * sizeof(float));

    // 1) precision conversion / padding
    {
        int n = BATCH * SEQ * EP + XPAD;
        convert_x<<<(n + 255) / 256, 256, 0, stream>>>(x, xh);
    }
    convert_w<<<(OC * 3 * EP + 255) / 256, 256, 0, stream>>>(W3, wh3, 3);
    convert_w<<<(OC * 4 * EP + 255) / 256, 256, 0, stream>>>(W4, wh4, 4);
    convert_w<<<(OC * 5 * EP + 255) / 256, 256, 0, stream>>>(W5, wh5, 5);

    // 2) fused conv-GEMM + ReLU + max-pool (the 59-GFLOP part, on WMMA)
    {
        dim3 grid(OC / 32, BATCH, 3);
        conv_gemm_max<<<grid, 128, 0, stream>>>(xh, wh3, wh4, wh5, b3, b4, b5, feat);
    }

    // 3) hierarchical softmax head
    head_kernel<<<BATCH, 128, 0, stream>>>(feat, targets, W_root, b_root,
                                           W_child, b_child,
                                           out_loss, out_root, out_child);
}
